// TopKRouter_1623497637919
// MI455X (gfx1250) — compile-verified, matrix operand order checked by reading, not run on hardware
//
#include <hip/hip_runtime.h>
#include <math.h>

#define INPUT_DIM   4096
#define NUM_EXPERTS 64
#define TOP_K       8
#define ROWS_PER_WAVE 16
#define K_SPLIT     2
#define K_HALF      (INPUT_DIM / K_SPLIT)                  // 2048
#define STRIPS_PER_BLOCK 2
#define WAVES_PER_BLOCK (STRIPS_PER_BLOCK * K_SPLIT)       // 4
#define ROWS_PER_BLOCK (ROWS_PER_WAVE * STRIPS_PER_BLOCK)  // 32
#define LDS_STRIDE  (NUM_EXPERTS + 1)                      // 65, conflict-free

typedef __attribute__((ext_vector_type(16))) __bf16 v16bf;
typedef __attribute__((ext_vector_type(8)))  float  v8f;

// ---------------------------------------------------------------------------
// Kernel 1: convert router weights fp32 -> bf16 into workspace (512 KB).
// ---------------------------------------------------------------------------
__global__ void topk_router_cvt_w(const float* __restrict__ W,
                                  __bf16* __restrict__ Wb, int n) {
    int i = blockIdx.x * blockDim.x + threadIdx.x;
    if (i < n) Wb[i] = (__bf16)W[i];
}

// ---------------------------------------------------------------------------
// Kernel 2: fused GEMM (bf16 WMMA, fp32 accum, split-K=2) + bias + top-8 +
// softmax. Each wave computes a 16x64 partial-score tile over half of K via
// v_wmma_f32_16x16x32_bf16; partials merged through LDS.
// ---------------------------------------------------------------------------
__device__ __forceinline__ v16bf cvt16(const float* p) {
    float4 f0 = *(const float4*)(p + 0);
    float4 f1 = *(const float4*)(p + 4);
    float4 f2 = *(const float4*)(p + 8);
    float4 f3 = *(const float4*)(p + 12);
    v16bf r;
    r[0]  = (__bf16)f0.x; r[1]  = (__bf16)f0.y; r[2]  = (__bf16)f0.z; r[3]  = (__bf16)f0.w;
    r[4]  = (__bf16)f1.x; r[5]  = (__bf16)f1.y; r[6]  = (__bf16)f1.z; r[7]  = (__bf16)f1.w;
    r[8]  = (__bf16)f2.x; r[9]  = (__bf16)f2.y; r[10] = (__bf16)f2.z; r[11] = (__bf16)f2.w;
    r[12] = (__bf16)f3.x; r[13] = (__bf16)f3.y; r[14] = (__bf16)f3.z; r[15] = (__bf16)f3.w;
    return r;
}

__global__ void __launch_bounds__(WAVES_PER_BLOCK * 32)
topk_router_main(const float* __restrict__ X,
                 const __bf16* __restrict__ Wb,
                 const float* __restrict__ bias,
                 float* __restrict__ probs_out,   // [rows, 8]
                 float* __restrict__ idx_out,     // [rows, 8] (indices as float)
                 int rows) {
    __shared__ float lds[K_SPLIT][ROWS_PER_BLOCK * LDS_STRIDE];

    const int tid   = threadIdx.x;
    const int wave  = tid >> 5;         // 0..3
    const int lane  = tid & 31;
    const int strip = wave & 1;         // row strip within block
    const int kh    = wave >> 1;        // K half (0 or 1)
    const int ll    = lane & 15;        // lane % 16
    const int lh    = lane >> 4;        // 0 or 1
    const int rowBase = blockIdx.x * ROWS_PER_BLOCK + strip * ROWS_PER_WAVE;
    const int kBase   = kh * K_HALF;

    // A frag: lane holds row (rowBase + ll), 16 contiguous K at (kBase + lh*16)
    const float* aRow = X + (size_t)(rowBase + ll) * INPUT_DIM + kBase + lh * 16;

    // B frag n: lane holds expert (n*16 + ll), 16 contiguous K at (kBase + lh*16)
    const __bf16* bRow[4];
#pragma unroll
    for (int n = 0; n < 4; ++n)
        bRow[n] = Wb + (size_t)(n * 16 + ll) * INPUT_DIM + kBase + lh * 16;

    v8f acc0 = {0,0,0,0,0,0,0,0};
    v8f acc[4] = {acc0, acc0, acc0, acc0};

    for (int k0 = 0; k0 < K_HALF; k0 += 32) {
        v16bf a  = cvt16(aRow + k0);
        v16bf b0 = *(const v16bf*)(bRow[0] + k0);
        v16bf b1 = *(const v16bf*)(bRow[1] + k0);
        v16bf b2 = *(const v16bf*)(bRow[2] + k0);
        v16bf b3 = *(const v16bf*)(bRow[3] + k0);
        acc[0] = __builtin_amdgcn_wmma_f32_16x16x32_bf16(false, a, false, b0,
                                                         (short)0, acc[0], false, false);
        acc[1] = __builtin_amdgcn_wmma_f32_16x16x32_bf16(false, a, false, b1,
                                                         (short)0, acc[1], false, false);
        acc[2] = __builtin_amdgcn_wmma_f32_16x16x32_bf16(false, a, false, b2,
                                                         (short)0, acc[2], false, false);
        acc[3] = __builtin_amdgcn_wmma_f32_16x16x32_bf16(false, a, false, b3,
                                                         (short)0, acc[3], false, false);
    }

    // C layout: vgpr r -> local row (r + 8*lh), col (n*16 + ll).
    // kh==0 wave folds in the bias; kh==1 wave stores the bare partial.
#pragma unroll
    for (int n = 0; n < 4; ++n) {
        float bn = (kh == 0) ? bias[n * 16 + ll] : 0.0f;
#pragma unroll
        for (int r = 0; r < 8; ++r) {
            lds[kh][(strip * 16 + r + 8 * lh) * LDS_STRIDE + n * 16 + ll] =
                acc[n][r] + bn;
        }
    }

    __syncthreads();

    // Epilogue: one thread per row -> merge partials, top-8 (sorted desc,
    // lowest index on ties), softmax.
    if (tid < ROWS_PER_BLOCK) {
        const int row = blockIdx.x * ROWS_PER_BLOCK + tid;
        const float* s0 = &lds[0][tid * LDS_STRIDE];
        const float* s1 = &lds[1][tid * LDS_STRIDE];

        float v[TOP_K];
        int   ix[TOP_K];
#pragma unroll
        for (int j = 0; j < TOP_K; ++j) { v[j] = -INFINITY; ix[j] = 0; }

        for (int e = 0; e < NUM_EXPERTS; ++e) {
            float sv = s0[e] + s1[e];
            if (sv > v[TOP_K - 1]) {
                int p = TOP_K - 1;
                while (p > 0 && sv > v[p - 1]) {
                    v[p] = v[p - 1]; ix[p] = ix[p - 1]; --p;
                }
                v[p] = sv; ix[p] = e;
            }
        }

        float m = v[0];
        float sum = 0.f;
        float ex[TOP_K];
#pragma unroll
        for (int j = 0; j < TOP_K; ++j) { ex[j] = __expf(v[j] - m); sum += ex[j]; }
        float inv = 1.0f / sum;

#pragma unroll
        for (int j = 0; j < TOP_K; ++j) {
            probs_out[(size_t)row * TOP_K + j] = ex[j] * inv;
            idx_out[(size_t)row * TOP_K + j]   = (float)ix[j];
        }
    }
}

// ---------------------------------------------------------------------------
extern "C" void kernel_launch(void* const* d_in, const int* in_sizes, int n_in,
                              void* d_out, int out_size, void* d_ws, size_t ws_size,
                              hipStream_t stream) {
    const float* X    = (const float*)d_in[0];
    const float* W    = (const float*)d_in[1];
    const float* bias = (const float*)d_in[2];

    const int rows = in_sizes[0] / INPUT_DIM;     // 16384
    const int wElems = NUM_EXPERTS * INPUT_DIM;   // 262144 -> 512 KB bf16 in d_ws

    __bf16* Wb = (__bf16*)d_ws;

    // 1) W fp32 -> bf16 (one-shot, tiny)
    topk_router_cvt_w<<<(wElems + 255) / 256, 256, 0, stream>>>(W, Wb, wElems);

    // 2) fused WMMA GEMM (split-K=2) + top-k + softmax
    float* probs = (float*)d_out;
    float* idxf  = (float*)d_out + (size_t)rows * TOP_K;
    int blocks = rows / ROWS_PER_BLOCK;           // 512
    topk_router_main<<<blocks, WAVES_PER_BLOCK * 32, 0, stream>>>(
        X, Wb, bias, probs, idxf, rows);
}
